// DeformableAttnBlock_MMA_2860448219495
// MI455X (gfx1250) — compile-verified, matrix operand order checked by reading, not run on hardware
//
#include <hip/hip_runtime.h>

typedef __attribute__((ext_vector_type(16))) __bf16 v16bf;
typedef __attribute__((ext_vector_type(8)))  float  v8f;

#define HH 32
#define WW 32
#define HW 1024

union V16 { v16bf v; uint4 q[2]; };

static __device__ inline unsigned short f2bf(float f) {
  unsigned u = __float_as_uint(f);
  unsigned r = u + 0x7FFFu + ((u >> 16) & 1u);   // round-to-nearest-even
  return (unsigned short)(r >> 16);
}

// ---------------------------------------------------------------------------
// Elementwise f32 -> bf16 (also used for weight prep: [co][ci][3][3] == [co][K])
// ---------------------------------------------------------------------------
__global__ void f32_to_bf16_kernel(const float* __restrict__ a,
                                   unsigned short* __restrict__ b, int n) {
  int i = blockIdx.x * 256 + threadIdx.x;
  if (i < n) b[i] = f2bf(a[i]);
}

// ---------------------------------------------------------------------------
// flow_out = base + bilinear(src, grid + base)   (align_corners=True, zeros pad)
// base/src/out: [2][HW] f32
// ---------------------------------------------------------------------------
__global__ void flow_compose_kernel(const float* __restrict__ base,
                                    const float* __restrict__ src,
                                    float* __restrict__ out) {
  int pix = blockIdx.x * 256 + threadIdx.x;
  if (pix >= HW) return;
  int py = pix >> 5, px = pix & 31;
  float bx = base[pix], by = base[HW + pix];
  float sx = px + bx, sy = py + by;
  float x0f = floorf(sx), y0f = floorf(sy);
  int x0 = (int)x0f, y0 = (int)y0f;
  float wx = sx - x0f, wy = sy - y0f;
  float s0 = 0.f, s1 = 0.f;
  for (int b = 0; b < 2; ++b) {
    int Y = y0 + b; if (Y < 0 || Y >= HH) continue;
    float wyv = b ? wy : (1.f - wy);
    for (int a = 0; a < 2; ++a) {
      int X = x0 + a; if (X < 0 || X >= WW) continue;
      float wv = wyv * (a ? wx : (1.f - wx));
      int idx = Y * WW + X;
      s0 += wv * src[idx];
      s1 += wv * src[HW + idx];
    }
  }
  out[pix] = bx + s0;
  out[HW + pix] = by + s1;
}

// ---------------------------------------------------------------------------
// Build extra tensor [460][HW] in bf16:
//   [x(:,0) | 6 flow-warped frames | 6 flows(2ch each)]
// ---------------------------------------------------------------------------
__global__ void build_extra_kernel(const float* __restrict__ x,
                                   const float* f0, const float* f1,
                                   const float* f2, const float* f3,
                                   const float* f4, const float* f5,
                                   unsigned short* __restrict__ extra) {
  int pix = blockIdx.x * 256 + threadIdx.x;
  if (pix >= HW) return;
  int py = pix >> 5, px = pix & 31;
  for (int c = 0; c < 64; ++c)
    extra[(size_t)c * HW + pix] = f2bf(x[(size_t)c * HW + pix]);
  const float* flows[6] = { f0, f1, f2, f3, f4, f5 };
  const int srcf[6] = { 1, 2, 0, 2, 1, 0 };
  for (int w = 0; w < 6; ++w) {
    const float* fl = flows[w];
    float sx = px + fl[pix], sy = py + fl[HW + pix];
    float x0f = floorf(sx), y0f = floorf(sy);
    int x0 = (int)x0f, y0 = (int)y0f;
    float wx = sx - x0f, wy = sy - y0f;
    const float* img = x + (size_t)srcf[w] * 64 * HW;
    for (int c = 0; c < 64; ++c) {
      float s = 0.f;
      for (int b = 0; b < 2; ++b) {
        int Y = y0 + b; if (Y < 0 || Y >= HH) continue;
        float wyv = b ? wy : (1.f - wy);
        for (int a = 0; a < 2; ++a) {
          int X = x0 + a; if (X < 0 || X >= WW) continue;
          s += wyv * (a ? wx : (1.f - wx)) * img[(size_t)c * HW + Y * WW + X];
        }
      }
      extra[(size_t)(64 + w * 64 + c) * HW + pix] = f2bf(s);
    }
  }
  for (int j = 0; j < 6; ++j) {
    extra[(size_t)(448 + j * 2 + 0) * HW + pix] = f2bf(flows[j][pix]);
    extra[(size_t)(448 + j * 2 + 1) * HW + pix] = f2bf(flows[j][HW + pix]);
  }
}

// ---------------------------------------------------------------------------
// Implicit-im2col 3x3 SAME conv as WMMA GEMM.
//   M = 1024 pixels, N = Cout, K = Cin*9.
//   Block: 128 pix x 64 cout, 4 waves; each wave: 32 pix x 64 cout (8 WMMA/Kstep).
//   Weights pre-laid-out [Cout][K] bf16, so the B tile async-copies into LDS
//   already K-transposed ([cout][kk]) via GLOBAL_LOAD_ASYNC_TO_LDS_B128.
//   A tile is im2col-gathered with zero padding (borders + K tail).
//   Fragments read from LDS as 2x ds_load_b128 per lane, matching the CDNA5
//   16-bit 16x32 A layout (lanes 0-15: K 0-7/16-23, lanes 16-31: K 8-15/24-31).
// ---------------------------------------------------------------------------
__global__ __launch_bounds__(128)
void conv_wmma_kernel(const unsigned short* __restrict__ in,   // [Cin][HW] bf16
                      const unsigned short* __restrict__ wtr,  // [Cout][K] bf16
                      const float* __restrict__ bias,          // [Cout]
                      float* __restrict__ outF,                // [Cout][HW] or null
                      unsigned short* __restrict__ outB,       // [Cout][HW] or null
                      int Cin, int Cout, int doLrelu) {
  const int K    = Cin * 9;
  const int nNT  = (Cout + 63) >> 6;
  const int mt   = blockIdx.x / nNT;
  const int nt   = blockIdx.x - mt * nNT;
  const int pix0 = mt << 7;   // 128-pixel tile
  const int n0   = nt << 6;   // 64-cout tile
  const int tid  = threadIdx.x;
  const int lane = tid & 31;
  const int wid  = tid >> 5;

  __shared__ __align__(16) unsigned short Alds[128][32];  // [pixel][k]
  __shared__ __align__(16) unsigned short Blds[64][32];   // [cout][k] (K-transposed)

  v8f accL[4], accH[4];
#pragma unroll
  for (int j = 0; j < 4; ++j) {
    v8f z = {0.f, 0.f, 0.f, 0.f, 0.f, 0.f, 0.f, 0.f};
    accL[j] = z;
    accH[j] = z;
  }

  // per-thread B-tile chunk geometry (fixed across K steps)
  const int brow0 = tid >> 2;               // chunk tid      -> row 0..31
  const int bcol0 = (tid & 3) << 3;         //                 col {0,8,16,24}
  const int brow1 = (tid + 128) >> 2;       // chunk tid+128  -> row 32..63
  const int bcol1 = bcol0;
  const unsigned lds_b0 = (unsigned)(size_t)&Blds[brow0][bcol0];
  const unsigned lds_b1 = (unsigned)(size_t)&Blds[brow1][bcol1];

  const int kTiles = (K + 31) >> 5;
  for (int kt = 0; kt < kTiles; ++kt) {
    const int k0 = kt << 5;

    // ---- B tile: async copy global -> LDS (contiguous b128 chunks) ----
    {
      unsigned long long ga0 =
          (unsigned long long)(size_t)(wtr + (size_t)(n0 + brow0) * K + (k0 + bcol0));
      unsigned long long ga1 =
          (unsigned long long)(size_t)(wtr + (size_t)(n0 + brow1) * K + (k0 + bcol1));
      asm volatile("global_load_async_to_lds_b128 %0, %1, off"
                   :: "v"(lds_b0), "v"(ga0) : "memory");
      asm volatile("global_load_async_to_lds_b128 %0, %1, off"
                   :: "v"(lds_b1), "v"(ga1) : "memory");
    }

    // ---- A tile: im2col gather with zero pad (borders + K tail) ----
    for (int i = tid; i < 4096; i += 128) {
      const int pp = i >> 5;
      const int kk = i & 31;
      const int k  = k0 + kk;
      unsigned short av = 0;
      if (k < K) {
        int ci = k / 9;
        int r  = k - ci * 9;
        int ky = r / 3;
        int kx = r - ky * 3;
        int pix = pix0 + pp;
        int sy = (pix >> 5) + ky - 1;
        int sx = (pix & 31) + kx - 1;
        if (sx >= 0 && sx < WW && sy >= 0 && sy < HH)
          av = in[(size_t)ci * HW + sy * WW + sx];
      }
      Alds[pp][kk] = av;
    }

    asm volatile("s_wait_asynccnt 0x0" ::: "memory");
    __syncthreads();

    // ---- fragments + 8 WMMA ----
    const int c0 = lane >> 4;  // K chunk select per CDNA5 16-bit layout
    V16 a0, a1;
    {
      const uint4* pr0 = (const uint4*)(&Alds[(wid << 5) + (lane & 15)][0]);
      a0.q[0] = pr0[c0];
      a0.q[1] = pr0[2 + c0];
      const uint4* pr1 = (const uint4*)(&Alds[(wid << 5) + 16 + (lane & 15)][0]);
      a1.q[0] = pr1[c0];
      a1.q[1] = pr1[2 + c0];
    }
#pragma unroll
    for (int j = 0; j < 4; ++j) {
      V16 b;
      const uint4* pr = (const uint4*)(&Blds[(j << 4) + (lane & 15)][0]);
      b.q[0] = pr[c0];
      b.q[1] = pr[2 + c0];
      accL[j] = __builtin_amdgcn_wmma_f32_16x16x32_bf16(
          false, a0.v, false, b.v, (short)0, accL[j], false, false);
      accH[j] = __builtin_amdgcn_wmma_f32_16x16x32_bf16(
          false, a1.v, false, b.v, (short)0, accH[j], false, false);
    }
    __syncthreads();
  }

  // ---- epilogue: bias + leaky-ReLU, f32/bf16 stores ----
  const int nl   = lane & 15;
  const int mrow = (lane >> 4) << 3;
#pragma unroll
  for (int j = 0; j < 4; ++j) {
    int co = n0 + (j << 4) + nl;
    if (co >= Cout) continue;
    float bvv = bias[co];
#pragma unroll
    for (int e = 0; e < 8; ++e) {
      float rL = accL[j][e] + bvv;
      float rH = accH[j][e] + bvv;
      if (doLrelu) {
        rL = (rL >= 0.f) ? rL : 0.1f * rL;
        rH = (rH >= 0.f) ? rH : 0.1f * rH;
      }
      int pixL = pix0 + (wid << 5) + mrow + e;
      int pixH = pixL + 16;
      if (outF) {
        outF[(size_t)co * HW + pixL] = rL;
        outF[(size_t)co * HW + pixH] = rH;
      }
      if (outB) {
        outB[(size_t)co * HW + pixL] = f2bf(rL);
        outB[(size_t)co * HW + pixH] = f2bf(rH);
      }
    }
  }
}

// ---------------------------------------------------------------------------
// In-place softmax over 324 attention logits per (query, head).
// aw layout: [7776][HW] f32, channel = t*2592 + m*324 + j.  One wave per (q,m).
// ---------------------------------------------------------------------------
__global__ __launch_bounds__(256) void softmax_aw_kernel(float* __restrict__ aw) {
  int wgid = blockIdx.x * 8 + (threadIdx.x >> 5);
  int lane = threadIdx.x & 31;
  int q = wgid >> 3, m = wgid & 7;
  int t = q >> 10, pix = q & 1023;
  size_t base = (size_t)(t * 2592 + m * 324) * HW + pix;
  float v[11];
  float mx = -3.0e38f;
#pragma unroll
  for (int i = 0; i < 11; ++i) {
    int j = lane + i * 32;
    v[i] = (j < 324) ? aw[base + (size_t)j * HW] : -3.0e38f;
    mx = fmaxf(mx, v[i]);
  }
  for (int o = 16; o > 0; o >>= 1) mx = fmaxf(mx, __shfl_xor(mx, o, 32));
  float s = 0.f;
#pragma unroll
  for (int i = 0; i < 11; ++i) {
    int j = lane + i * 32;
    if (j < 324) { v[i] = __expf(v[i] - mx); s += v[i]; }
  }
  for (int o = 16; o > 0; o >>= 1) s += __shfl_xor(s, o, 32);
  float inv = 1.f / s;
#pragma unroll
  for (int i = 0; i < 11; ++i) {
    int j = lane + i * 32;
    if (j < 324) aw[base + (size_t)j * HW] = v[i] * inv;
  }
}

// ---------------------------------------------------------------------------
// Deformable attention: one wave per (query q, head m); lanes stride the 324
// (level, point, patch) samples; 4-tap zero-pad bilinear over D=8 channels.
// ---------------------------------------------------------------------------
__global__ __launch_bounds__(128)
void deform_attn_kernel(const float* __restrict__ value,  // [192][HW]
                        const float* __restrict__ so,     // [1728][HW]
                        const float* __restrict__ aw,     // [7776][HW] softmaxed
                        const float* fcn1, const float* fcn2, const float* fn1c,
                        const float* fn1n2, const float* fn2n1, const float* fn2c,
                        unsigned short* __restrict__ outB)  // [192][HW] bf16
{
  int wgid = blockIdx.x * 4 + (threadIdx.x >> 5);
  int lane = threadIdx.x & 31;
  int q = wgid >> 3, m = wgid & 7;
  int t = q >> 10, pix = q & 1023;
  int py = pix >> 5, px = pix & 31;
  const float* ftab[9] = { nullptr, fcn1, fcn2, fn1c, nullptr, fn1n2, fn2c, fn2n1, nullptr };
  float acc[8];
#pragma unroll
  for (int d = 0; d < 8; ++d) acc[d] = 0.f;

  for (int j = lane; j < 324; j += 32) {
    int l = j / 108;
    int rem = j - l * 108;
    int p  = rem / 9;
    int pt = rem - p * 9;
    int soch = t * 576 + ((m * 3 + l) * 12 + p) * 2;
    float ox = so[(size_t)soch * HW + pix];
    float oy = so[(size_t)(soch + 1) * HW + pix];
    const float* fp = ftab[t * 3 + l];
    float fx = fp ? fp[pix] : 0.f;
    float fy = fp ? fp[HW + pix] : 0.f;
    float sx = px + ox + fx + (float)(pt / 3 - 1);
    float sy = py + oy + fy + (float)(pt - (pt / 3) * 3 - 1);
    float wq = aw[(size_t)(t * 2592 + m * 324 + j) * HW + pix];
    float x0f = floorf(sx), y0f = floorf(sy);
    int x0 = (int)x0f, y0 = (int)y0f;
    float wx = sx - x0f, wy = sy - y0f;
    const float* vb = value + (size_t)(l * 64 + m * 8) * HW;
#pragma unroll
    for (int b = 0; b < 2; ++b) {
      int Y = y0 + b; if (Y < 0 || Y >= HH) continue;
      float wyv = b ? wy : (1.f - wy);
#pragma unroll
      for (int a = 0; a < 2; ++a) {
        int X = x0 + a; if (X < 0 || X >= WW) continue;
        float wv = wq * wyv * (a ? wx : (1.f - wx));
        int idx = Y * WW + X;
#pragma unroll
        for (int d = 0; d < 8; ++d) acc[d] += wv * vb[(size_t)d * HW + idx];
      }
    }
  }
#pragma unroll
  for (int d = 0; d < 8; ++d)
    for (int o = 16; o > 0; o >>= 1) acc[d] += __shfl_down(acc[d], o, 32);
  if (lane == 0) {
#pragma unroll
    for (int d = 0; d < 8; ++d)
      outB[(size_t)(t * 64 + m * 8 + d) * HW + pix] = f2bf(acc[d]);
  }
}

// ---------------------------------------------------------------------------
// d_out = op1_out + x
// ---------------------------------------------------------------------------
__global__ void residual_add_kernel(const float* __restrict__ a,
                                    const float* __restrict__ x,
                                    float* __restrict__ o, int n) {
  int i = blockIdx.x * 256 + threadIdx.x;
  if (i < n) o[i] = a[i] + x[i];
}

// ---------------------------------------------------------------------------
extern "C" void kernel_launch(void* const* d_in, const int* in_sizes, int n_in,
                              void* d_out, int out_size, void* d_ws, size_t ws_size,
                              hipStream_t stream) {
  (void)in_sizes; (void)n_in; (void)out_size; (void)ws_size;
  const float* x     = (const float*)d_in[0];
  const float* flow1 = (const float*)d_in[1];
  const float* flow2 = (const float*)d_in[2];
  const float* flip1 = (const float*)d_in[3];
  const float* flip2 = (const float*)d_in[4];
  const float* W[12]; const float* B[12];
  for (int i = 0; i < 12; ++i) {
    W[i] = (const float*)d_in[5 + 2 * i];
    B[i] = (const float*)d_in[6 + 2 * i];
  }
  //               vp0   vp1   so0  so1 so2 so3   aw0  aw1 aw2 aw3   op0   op1
  const int KK[12] = {1728, 1728, 4140, 576, 576, 576, 4140, 576, 576, 576, 1728, 1728};
  const int CO[12] = {192,  192,  64,   64,  64,  1728, 64,  64,  64,  7776, 192,  192};
  const int CI[12] = {192,  192,  460,  64,  64,  64,  460,  64,  64,  64,  192,  192};

  size_t off = 0;
  auto alloc = [&](size_t b) -> void* {
    off = (off + 255) & ~(size_t)255;
    void* p = (char*)d_ws + off;
    off += b;
    return p;
  };
  unsigned short* xb    = (unsigned short*)alloc((size_t)196608 * 2);
  unsigned short* extra = (unsigned short*)alloc((size_t)460 * HW * 2);
  float* fcn2 = (float*)alloc(2048 * 4);
  float* fn2c = (float*)alloc(2048 * 4);
  unsigned short* wb[12];
  for (int i = 0; i < 12; ++i) wb[i] = (unsigned short*)alloc((size_t)KK[i] * CO[i] * 2);
  unsigned short* act0  = (unsigned short*)alloc((size_t)192 * HW * 2);
  unsigned short* act1  = (unsigned short*)alloc((size_t)192 * HW * 2);
  float* value = (float*)alloc((size_t)192 * HW * 4);
  float* soO   = (float*)alloc((size_t)1728 * HW * 4);
  float* awO   = (float*)alloc((size_t)7776 * HW * 4);
  unsigned short* attnb = (unsigned short*)alloc((size_t)192 * HW * 2);
  float* opO   = (float*)alloc((size_t)192 * HW * 4);

  // prep
  f32_to_bf16_kernel<<<768, 256, 0, stream>>>(x, xb, 196608);
  flow_compose_kernel<<<4, 256, 0, stream>>>(flow1, flow2, fcn2);   // flow_cn2
  flow_compose_kernel<<<4, 256, 0, stream>>>(flip2, flip1, fn2c);   // flow_n2c
  build_extra_kernel<<<4, 256, 0, stream>>>(x, flow1, fcn2, flip1, flow2, flip2, fn2c, extra);
  for (int i = 0; i < 12; ++i) {
    int n = KK[i] * CO[i];   // weight layout [co][K] is the native layout: plain convert
    f32_to_bf16_kernel<<<(n + 255) / 256, 256, 0, stream>>>(W[i], wb[i], n);
  }

  auto conv = [&](const unsigned short* in, int idx, float* of, unsigned short* ob, int act) {
    int nNT = (CO[idx] + 63) / 64;
    conv_wmma_kernel<<<8 * nNT, 128, 0, stream>>>(in, wb[idx], B[idx], of, ob,
                                                  CI[idx], CO[idx], act);
  };
  conv(xb,    0, nullptr, act0, 1);   // vp0 + lrelu
  conv(act0,  1, value,  nullptr, 0); // vp1 -> value (f32)
  conv(extra, 2, nullptr, act0, 1);   // so0
  conv(act0,  3, nullptr, act1, 1);   // so1
  conv(act1,  4, nullptr, act0, 1);   // so2
  conv(act0,  5, soO,    nullptr, 0); // so3 -> offsets
  conv(extra, 6, nullptr, act1, 1);   // aw0
  conv(act1,  7, nullptr, act0, 1);   // aw1
  conv(act0,  8, nullptr, act1, 1);   // aw2
  conv(act1,  9, awO,    nullptr, 0); // aw3 -> attention logits

  softmax_aw_kernel<<<3072, 256, 0, stream>>>(awO);
  deform_attn_kernel<<<6144, 128, 0, stream>>>(value, soO, awO,
                                               flow1, fcn2, flip1, flow2, flip2, fn2c,
                                               attnb);
  conv(attnb, 10, nullptr, act0, 1);  // op0 + lrelu
  conv(act0,  11, opO,   nullptr, 0); // op1
  residual_add_kernel<<<768, 256, 0, stream>>>(opO, x, (float*)d_out, 196608);
}
